// QuantumMixedStateAttention_19576460935328
// MI455X (gfx1250) — matrix operand order repaired and checked by reading, not compiled, not run on hardware
//
#include <hip/hip_runtime.h>
#include <hip/hip_bf16.h>
#include <math.h>
#include <stdint.h>

// ---------------------------------------------------------------------------
// QuantumMixedStateAttention for gfx1250 (MI455X), wave32, WMMA bf16/f32,
// TDM async tile loads where available.
// B=4, S=2048, D=768, Q=8 qubits, H=12 heads, Dh=64.
// ---------------------------------------------------------------------------

#define B_  4
#define S_  2048
#define D_  768
#define H_  12
#define DH_ 64
#define QB_ 8
#define ROWS_ (B_ * S_)          // 8192
#define D3_  (3 * D_)            // 2304

typedef __attribute__((ext_vector_type(8)))  float   v8f;
typedef __attribute__((ext_vector_type(8)))  __bf16  v8bf;
typedef __attribute__((ext_vector_type(16))) __bf16  v16bf;

#if defined(__has_builtin)
#  if __has_builtin(__builtin_amdgcn_tensor_load_to_lds) && \
      __has_builtin(__builtin_amdgcn_s_wait_tensorcnt)
#    define HAVE_TDM 1
#  endif
#  if __has_builtin(__builtin_amdgcn_rcpf)
#    define FAST_RCP(x) __builtin_amdgcn_rcpf(x)
#  endif
#endif
#ifndef HAVE_TDM
#  define HAVE_TDM 0
#endif
#ifndef FAST_RCP
#  define FAST_RCP(x) (1.0f / (x))
#endif

union ABf { v16bf v; v8bf h[2]; };

static __device__ __forceinline__ v8f wmma_bf16(v16bf a, v16bf b, v8f c) {
  // D = A(16x32 bf16) * B(32x16 bf16) + C(16x16 f32)
  return __builtin_amdgcn_wmma_f32_16x16x32_bf16(false, a, false, b, (short)0, c,
                                                 false, false);
}

// A-operand fragment (16x32 bf16): lane L<16 -> M=L, K [0..15];
// lane L>=16 -> M=L-16, K [16..31].  Two 16B loads.
static __device__ __forceinline__ v16bf load_a(const __bf16* rowptr, int lane) {
  int ka = (lane < 16) ? 0 : 16;
  ABf u;
  u.h[0] = *(const v8bf*)(rowptr + ka);
  u.h[1] = *(const v8bf*)(rowptr + ka + 8);
  return u.v;
}

#if HAVE_TDM
typedef __attribute__((ext_vector_type(4))) unsigned int v4u;
typedef __attribute__((ext_vector_type(8))) int v8i_;
typedef __attribute__((ext_vector_type(4))) int v4i_;

// 2D tile DMA: global (row-major, tile_d0 units wide, tile_d1 rows, row stride
// stride_units) -> LDS at byte offset lds_off.  data_size: 0=1B,1=2B,2=4B,3=8B.
static __device__ __forceinline__ void tdm_load_2d(unsigned lds_off, const void* gptr,
                                                   unsigned tile_d0, unsigned tile_d1,
                                                   unsigned stride_units,
                                                   unsigned data_size) {
  unsigned long long ga = (unsigned long long)(uintptr_t)gptr;
  // D# group0: count=1 | lds_addr | global_addr[56:0] | type=2
  v4u g0 = {1u, lds_off, (unsigned)(ga & 0xffffffffu),
            (unsigned)((ga >> 32) & 0x01ffffffu) | (2u << 30)};
  // tensor_dim0 = tensor_dim1 = 1<<20 (large: no OOB zero-fill in-tile)
  const unsigned td_hi16 = 0x10;   // (1<<20) >> 16
  v8i_ g1 = {(int)(data_size << 16),                 // wg_mask=0, data_size
             (int)(0u << 16),                        // atomic_addr=0, td0[15:0]=0
             (int)((0u << 16) | td_hi16),            // td0[31:16], td1[15:0]=0
             (int)((tile_d0 << 16) | td_hi16),       // td1[31:16], tile_dim0
             (int)(tile_d1 & 0xffffu),               // tile_dim1, tile_dim2=0
             (int)stride_units,                      // tensor_dim0_stride[31:0]
             0, 0};                                  // stride hi, dim1_stride=0
  v4i_ gz = {0, 0, 0, 0};
#if __clang_major__ >= 23
  v8i_ gz8 = {0, 0, 0, 0, 0, 0, 0, 0};
  __builtin_amdgcn_tensor_load_to_lds(g0, g1, gz, gz, gz8, 0);
#else
  __builtin_amdgcn_tensor_load_to_lds(g0, g1, gz, gz, 0);
#endif
}

static __device__ __forceinline__ unsigned lds_offset_of(const void* p) {
  // flat LDS aperture: low 32 bits == offset within wave's LDS allocation
  return (unsigned)(uintptr_t)p;
}
#endif  // HAVE_TDM

// ---------------------------------------------------------------------------
// 0) fp32 -> bf16 conversion (x, in_w, out_w)
// ---------------------------------------------------------------------------
__global__ void f2bf_kernel(const float* __restrict__ src, __bf16* __restrict__ dst, int n) {
  int i = blockIdx.x * 256 + threadIdx.x;
  if (i < n) dst[i] = (__bf16)src[i];
}

// ---------------------------------------------------------------------------
// 1) State projection + normalize: st[row][16] = (x@sp_w.T + sp_b)/(||.||+eps)
// ---------------------------------------------------------------------------
__global__ __launch_bounds__(256) void state_kernel(const float* __restrict__ x,
                                                    const float* __restrict__ sp_w,
                                                    const float* __restrict__ sp_b,
                                                    float* __restrict__ st) {
  __shared__ float wsh[16 * D_];
  for (int i = threadIdx.x; i < 16 * D_; i += 256) wsh[i] = sp_w[i];
  __syncthreads();
  int wv = threadIdx.x >> 5, lane = threadIdx.x & 31;
  int row = blockIdx.x * 8 + wv;
  const float* xr = x + (size_t)row * D_;
  float acc[16];
#pragma unroll
  for (int o = 0; o < 16; ++o) acc[o] = 0.f;
  for (int c = lane; c < D_; c += 32) {
    float xv = xr[c];
#pragma unroll
    for (int o = 0; o < 16; ++o) acc[o] += xv * wsh[o * D_ + c];
  }
#pragma unroll
  for (int o = 0; o < 16; ++o)
    for (int off = 16; off; off >>= 1) acc[o] += __shfl_xor(acc[o], off, 32);
  float n2 = 0.f;
#pragma unroll
  for (int o = 0; o < 16; ++o) { float t = acc[o] + sp_b[o]; acc[o] = t; n2 += t * t; }
  float inv = 1.f / (sqrtf(n2) + 1e-8f);
  float val = 0.f;
#pragma unroll
  for (int o = 0; o < 16; ++o) if (lane == o) val = acc[o];
  if (lane < 16) st[(size_t)row * 16 + lane] = val * inv;
}

// ---------------------------------------------------------------------------
// 2) Quantum softmax statistics (row max & sum of exp(fid)) — two-pass softmax.
// ---------------------------------------------------------------------------
__global__ __launch_bounds__(256) void qstats_kernel(const float* __restrict__ st,
                                                     float* __restrict__ qmax,
                                                     float* __restrict__ qsum) {
  int wv = threadIdx.x >> 5, lane = threadIdx.x & 31;
  int row = blockIdx.x * 8 + wv;          // 0..8191
  int bbase = (row >> 11) << 11;          // batch row base
  float si[16];
#pragma unroll
  for (int q = 0; q < 16; ++q) si[q] = st[(size_t)row * 16 + q];
  float m = -3.0e38f, s = 0.f;
  for (int jt = 0; jt < S_ / 32; ++jt) {
    int j = bbase + jt * 32 + lane;
    float sj[16];
#pragma unroll
    for (int q = 0; q < 16; ++q) sj[q] = st[(size_t)j * 16 + q];
    float ovre = 0.f, ovim = 0.f;
#pragma unroll
    for (int q = 0; q < QB_; ++q) {
      ovre += si[q] * sj[q] + si[q + 8] * sj[q + 8];
      ovim += si[q] * sj[q + 8] - si[q + 8] * sj[q];
    }
    float fid = ovre * ovre + ovim * ovim;
    float nm = fmaxf(m, fid);
    s = s * __expf(m - nm) + __expf(fid - nm);
    m = nm;
  }
  for (int off = 16; off; off >>= 1) {
    float om = __shfl_xor(m, off, 32), os = __shfl_xor(s, off, 32);
    float nm = fmaxf(m, om);
    s = s * __expf(m - nm) + os * __expf(om - nm);
    m = nm;
  }
  if (lane == 0) { qmax[row] = m; qsum[row] = s; }
}

// ---------------------------------------------------------------------------
// 3) quantum_out = softmax(fid) @ x, WMMA-tiled; writes sigmoid(qw)*result.
//    Block: 256 thr (8 waves). Tile: 32 rows x 256 cols, K(j)-steps of 32.
//    st j-tile fetched by TDM (wave 0) when available.
// ---------------------------------------------------------------------------
__global__ __launch_bounds__(256) void quantum_out_kernel(const float* __restrict__ st,
                                                          const float* __restrict__ qmax,
                                                          const float* __restrict__ qsum,
                                                          const __bf16* __restrict__ xb,
                                                          const float* __restrict__ qwp,
                                                          float* __restrict__ out) {
  __shared__ float  sti[32][16];
  __shared__ float  stj[32][16];
  __shared__ __bf16 pt[32][32];
  __shared__ __bf16 xT[256][32];   // [d][j], transposed so B-frags are contiguous in K
  int b = blockIdx.z, dc = blockIdx.x;
  int ibase = blockIdx.y * 32, dbase = dc * 256;
  int tid = threadIdx.x, lane = tid & 31, wv = tid >> 5;
  size_t brow = (size_t)b * S_;

  for (int e = tid; e < 512; e += 256) sti[e >> 4][e & 15] = st[(brow + ibase) * 16 + e];

  float mrow = qmax[brow + ibase + (tid & 31)];
  float sinv = FAST_RCP(qsum[brow + ibase + (tid & 31)]);

  v8f acc[2][2];
  v8f vzero = {0.f, 0.f, 0.f, 0.f, 0.f, 0.f, 0.f, 0.f};
#pragma unroll
  for (int m = 0; m < 2; ++m)
#pragma unroll
    for (int nn = 0; nn < 2; ++nn) acc[m][nn] = vzero;

  for (int jt = 0; jt < S_ / 32; ++jt) {
    int jbase = jt * 32;
    __syncthreads();                     // prior reads of stj/xT/pt complete
#if HAVE_TDM
    if (wv == 0)  // 32x16 f32 = 2KB contiguous: 256 x 8B units, 1 row
      tdm_load_2d(lds_offset_of(&stj[0][0]), st + (brow + jbase) * 16,
                  /*tile_d0=*/256, /*tile_d1=*/1, /*stride=*/256, /*ds=*/3);
#else
    for (int e = tid; e < 512; e += 256) stj[e >> 4][e & 15] = st[(brow + jbase) * 16 + e];
#endif
    {
      int jr = tid >> 3;              // 0..31
      int d0 = (tid & 7) * 32;        // 0..224
      const __bf16* src = xb + (brow + jbase + jr) * D_ + dbase + d0;
#pragma unroll
      for (int t = 0; t < 32; ++t) xT[d0 + t][jr] = src[t];
      if (jt + 1 < S_ / 32)
        __builtin_prefetch((const void*)(src + 32 * D_), 0, 1);
    }
#if HAVE_TDM
    if (wv == 0) __builtin_amdgcn_s_wait_tensorcnt(0);
#endif
    __syncthreads();
    // probabilities p[i][j] = exp(fid - max_i)/sum_i  (4 entries per thread)
#pragma unroll
    for (int e4 = 0; e4 < 4; ++e4) {
      int i = tid & 31;
      int j = e4 * 8 + (tid >> 5);
      float ovre = 0.f, ovim = 0.f;
#pragma unroll
      for (int q = 0; q < QB_; ++q) {
        ovre += sti[i][q] * stj[j][q] + sti[i][q + 8] * stj[j][q + 8];
        ovim += sti[i][q] * stj[j][q + 8] - sti[i][q + 8] * stj[j][q];
      }
      float fid = ovre * ovre + ovim * ovim;
      pt[i][j] = (__bf16)(__expf(fid - mrow) * sinv);
    }
    __syncthreads();
    // WMMA: acc[m][nn] += p(16x32) @ xT(32x16)
    v16bf a[2];
#pragma unroll
    for (int m = 0; m < 2; ++m) a[m] = load_a(&pt[m * 16 + (lane & 15)][0], lane);
    int kb = (lane < 16) ? 0 : 16;
#pragma unroll
    for (int nn = 0; nn < 2; ++nn) {
      int drow = (wv * 2 + nn) * 16 + (lane & 15);
      v16bf bf = *(const v16bf*)&xT[drow][kb];
#pragma unroll
      for (int m = 0; m < 2; ++m) acc[m][nn] = wmma_bf16(a[m], bf, acc[m][nn]);
    }
  }
  float sigw = FAST_RCP(1.f + __expf(-qwp[0]));
#pragma unroll
  for (int m = 0; m < 2; ++m)
#pragma unroll
    for (int nn = 0; nn < 2; ++nn) {
      int rbase = ibase + m * 16 + ((lane < 16) ? 0 : 8);
      int col = dbase + (wv * 2 + nn) * 16 + (lane & 15);
#pragma unroll
      for (int r = 0; r < 8; ++r)
        out[(brow + rbase + r) * D_ + col] = sigw * acc[m][nn][r];
    }
}

// ---------------------------------------------------------------------------
// 4) qkv = x @ in_w.T + in_b   (M=8192, N=2304, K=768), bf16 WMMA, 16x64/wave
// ---------------------------------------------------------------------------
__global__ __launch_bounds__(256) void qkv_gemm_kernel(const __bf16* __restrict__ xb,
                                                       const __bf16* __restrict__ wb,
                                                       const float* __restrict__ bias,
                                                       __bf16* __restrict__ qkv) {
  int tid = threadIdx.x, lane = tid & 31, wv = tid >> 5;
  int gw = blockIdx.x * 8 + wv;
  int Mi = gw / (D3_ / 64), Nj = gw % (D3_ / 64);
  const __bf16* arow = xb + (size_t)(Mi * 16 + (lane & 15)) * D_;
  int kb = (lane < 16) ? 0 : 16;
  v8f acc[4];
  v8f vzero = {0.f, 0.f, 0.f, 0.f, 0.f, 0.f, 0.f, 0.f};
#pragma unroll
  for (int n = 0; n < 4; ++n) acc[n] = vzero;
  for (int kk = 0; kk < D_ / 32; ++kk) {
    v16bf a = load_a(arow + kk * 32, lane);
#pragma unroll
    for (int n = 0; n < 4; ++n) {
      const __bf16* brow = wb + (size_t)(Nj * 64 + n * 16 + (lane & 15)) * D_;
      v16bf bf = *(const v16bf*)(brow + kk * 32 + kb);
      acc[n] = wmma_bf16(a, bf, acc[n]);
    }
  }
#pragma unroll
  for (int n = 0; n < 4; ++n) {
    int col = Nj * 64 + n * 16 + (lane & 15);
    float bv = bias[col];
    int rbase = Mi * 16 + ((lane < 16) ? 0 : 8);
#pragma unroll
    for (int r = 0; r < 8; ++r)
      qkv[(size_t)(rbase + r) * D3_ + col] = (__bf16)(acc[n][r] + bv);
  }
}

// ---------------------------------------------------------------------------
// 5) Classical flash attention per (b,h): 64-row i-tile, 4 waves, online
//    softmax.  K-tiles arrive via double-buffered TDM DMA when available.
// ---------------------------------------------------------------------------
__global__ __launch_bounds__(128) void attn_kernel(const __bf16* __restrict__ qkv,
                                                   __bf16* __restrict__ ctx) {
  __shared__ __bf16 ks[2][32][64];     // row-major k tiles [buf][j][d] (2x4KB)
  __shared__ __bf16 vT[64][32];        // [d][j]
  __shared__ __bf16 pb[4][16][32];     // per-wave p tile (M x K)
  int bh = blockIdx.y, b = bh / H_, h = bh % H_;
  int tid = threadIdx.x, lane = tid & 31, wv = tid >> 5;
  int ibase = blockIdx.x * 64 + wv * 16;
  size_t brow = (size_t)b * S_;
  int qoff = h * DH_, koff = D_ + h * DH_, voff = 2 * D_ + h * DH_;
  int kb = (lane < 16) ? 0 : 16;

  v16bf aq[2];
  {
    const __bf16* qrow = qkv + (brow + ibase + (lane & 15)) * D3_ + qoff;
#pragma unroll
    for (int kk = 0; kk < 2; ++kk) aq[kk] = load_a(qrow + kk * 32, lane);
  }
  v8f acc[4];
  v8f vzero = {0.f, 0.f, 0.f, 0.f, 0.f, 0.f, 0.f, 0.f};
#pragma unroll
  for (int nt = 0; nt < 4; ++nt) acc[nt] = vzero;
  float mrow[8], lrow[8];
#pragma unroll
  for (int r = 0; r < 8; ++r) { mrow[r] = -3.0e38f; lrow[r] = 0.f; }

#if HAVE_TDM
  if (wv == 0)  // prologue: DMA k-tile 0 (32 rows x 128B, row stride 4608B)
    tdm_load_2d(lds_offset_of(&ks[0][0][0]), qkv + brow * D3_ + koff,
                /*tile_d0=*/32, /*tile_d1=*/32, /*stride=*/D3_ * 2 / 4, /*ds=*/2);
#endif

  for (int jt = 0; jt < S_ / 32; ++jt) {
    int jbase = jt * 32;
    __syncthreads();                    // prev-iter ks/vT reads complete
#if HAVE_TDM
    if (wv == 0 && jt + 1 < S_ / 32)    // DMA next k-tile into other buffer
      tdm_load_2d(lds_offset_of(&ks[(jt + 1) & 1][0][0]),
                  qkv + (brow + jbase + 32) * D3_ + koff,
                  /*tile_d0=*/32, /*tile_d1=*/32, /*stride=*/D3_ * 2 / 4, /*ds=*/2);
#else
    {
      int jr = tid >> 2, d0 = (tid & 3) * 16;
      const __bf16* src = qkv + (brow + jbase + jr) * D3_ + koff + d0;
#pragma unroll
      for (int t = 0; t < 16; ++t) ks[0][jr][d0 + t] = src[t];
    }
#endif
    {
      int jr = tid >> 2;                // 0..31
      int d0 = (tid & 3) * 16;          // 0..48
      const __bf16* src = qkv + (brow + jbase + jr) * D3_ + voff + d0;
#pragma unroll
      for (int t = 0; t < 16; ++t) vT[d0 + t][jr] = src[t];
      if (jt + 1 < S_ / 32)
        __builtin_prefetch((const void*)(src + 32 * D3_), 0, 1);
    }
#if HAVE_TDM
    if (wv == 0) {                      // in-order TDM: tile jt complete
      if (jt + 1 < S_ / 32) __builtin_amdgcn_s_wait_tensorcnt(1);
      else                  __builtin_amdgcn_s_wait_tensorcnt(0);
    }
    const __bf16(*kst)[64] = ks[HAVE_TDM ? (jt & 1) : 0];
#else
    const __bf16(*kst)[64] = ks[0];
#endif
    __syncthreads();                    // ks/vT ready
    // scores (16 x 32) = q(16x64) @ k^T — B frags from LDS k tile
    v8f sc[2];
    sc[0] = vzero; sc[1] = vzero;
#pragma unroll
    for (int nt = 0; nt < 2; ++nt) {
      const __bf16* krow = &kst[nt * 16 + (lane & 15)][0];
#pragma unroll
      for (int kk = 0; kk < 2; ++kk) {
        v16bf bf = *(const v16bf*)(krow + kk * 32 + kb);
        sc[nt] = wmma_bf16(aq[kk], bf, sc[nt]);
      }
    }
    // online softmax per row (row = r + lane_half*8)
#pragma unroll
    for (int r = 0; r < 8; ++r) {
      float s0 = sc[0][r] * 0.125f, s1 = sc[1][r] * 0.125f;
      float t = fmaxf(s0, s1);
#pragma unroll
      for (int off = 8; off; off >>= 1) t = fmaxf(t, __shfl_xor(t, off, 32));
      float nm = fmaxf(mrow[r], t);
      float fsc = __expf(mrow[r] - nm);
      mrow[r] = nm;
      float p0 = __expf(s0 - nm), p1 = __expf(s1 - nm);
      float ps = p0 + p1;
#pragma unroll
      for (int off = 8; off; off >>= 1) ps += __shfl_xor(ps, off, 32);
      lrow[r] = lrow[r] * fsc + ps;
#pragma unroll
      for (int nt = 0; nt < 4; ++nt) acc[nt][r] *= fsc;
      int rr = r + ((lane < 16) ? 0 : 8);
      pb[wv][rr][lane & 15] = (__bf16)p0;
      pb[wv][rr][16 + (lane & 15)] = (__bf16)p1;
    }
    // ctx += p(16x32) @ v(32x64)
    v16bf ap = load_a(&pb[wv][lane & 15][0], lane);
#pragma unroll
    for (int nt = 0; nt < 4; ++nt) {
      v16bf bf = *(const v16bf*)&vT[nt * 16 + (lane & 15)][kb];
      acc[nt] = wmma_bf16(ap, bf, acc[nt]);
    }
  }
#pragma unroll
  for (int nt = 0; nt < 4; ++nt) {
    int col = h * DH_ + nt * 16 + (lane & 15);
#pragma unroll
    for (int r = 0; r < 8; ++r) {
      int row = ibase + r + ((lane < 16) ? 0 : 8);
      float li = FAST_RCP(lrow[r]);
      ctx[(brow + row) * D_ + col] = (__bf16)(acc[nt][r] * li);
    }
  }
}

// ---------------------------------------------------------------------------
// 6) out += (1-sigmoid(qw)) * (ctx @ out_w.T + out_b)   (fused blend)
// ---------------------------------------------------------------------------
__global__ __launch_bounds__(256) void outproj_kernel(const __bf16* __restrict__ ctxb,
                                                      const __bf16* __restrict__ wb,
                                                      const float* __restrict__ bias,
                                                      const float* __restrict__ qwp,
                                                      float* __restrict__ out) {
  int tid = threadIdx.x, lane = tid & 31, wv = tid >> 5;
  int gw = blockIdx.x * 8 + wv;
  int Mi = gw / (D_ / 64), Nj = gw % (D_ / 64);
  const __bf16* arow = ctxb + (size_t)(Mi * 16 + (lane & 15)) * D_;
  int kb = (lane < 16) ? 0 : 16;
  v8f acc[4];
  v8f vzero = {0.f, 0.f, 0.f, 0.f, 0.f, 0.f, 0.f, 0.f};
#pragma unroll
  for (int n = 0; n < 4; ++n) acc[n] = vzero;
  for (int kk = 0; kk < D_ / 32; ++kk) {
    v16bf a = load_a(arow + kk * 32, lane);
#pragma unroll
    for (int n = 0; n < 4; ++n) {
      const __bf16* brow = wb + (size_t)(Nj * 64 + n * 16 + (lane & 15)) * D_;
      v16bf bf = *(const v16bf*)(brow + kk * 32 + kb);
      acc[n] = wmma_bf16(a, bf, acc[n]);
    }
  }
  float cw = 1.f - FAST_RCP(1.f + __expf(-qwp[0]));
#pragma unroll
  for (int n = 0; n < 4; ++n) {
    int col = Nj * 64 + n * 16 + (lane & 15);
    float bv = bias[col];
    int rbase = Mi * 16 + ((lane < 16) ? 0 : 8);
#pragma unroll
    for (int r = 0; r < 8; ++r) {
      size_t idx = (size_t)(rbase + r) * D_ + col;
      out[idx] = out[idx] + cw * (acc[n][r] + bv);   // out holds sigw*quantum
    }
  }
}

// ---------------------------------------------------------------------------
extern "C" void kernel_launch(void* const* d_in, const int* in_sizes, int n_in,
                              void* d_out, int out_size, void* d_ws, size_t ws_size,
                              hipStream_t stream) {
  const float* x     = (const float*)d_in[0];
  const float* sp_w  = (const float*)d_in[1];
  const float* sp_b  = (const float*)d_in[2];
  const float* in_w  = (const float*)d_in[3];
  const float* in_b  = (const float*)d_in[4];
  const float* out_w = (const float*)d_in[5];
  const float* out_b = (const float*)d_in[6];
  const float* qw    = (const float*)d_in[7];
  float* out = (float*)d_out;

  char* p = (char*)d_ws;
  auto alloc = [&](size_t bytes) {
    void* r = (void*)p;
    p += (bytes + 255) & ~(size_t)255;
    return r;
  };
  __bf16* xb     = (__bf16*)alloc((size_t)ROWS_ * D_ * 2);
  __bf16* in_wb  = (__bf16*)alloc((size_t)D3_ * D_ * 2);
  __bf16* out_wb = (__bf16*)alloc((size_t)D_ * D_ * 2);
  float*  st     = (float*)alloc((size_t)ROWS_ * 16 * 4);
  float*  qmax   = (float*)alloc((size_t)ROWS_ * 4);
  float*  qsum   = (float*)alloc((size_t)ROWS_ * 4);
  __bf16* qkvb   = (__bf16*)alloc((size_t)ROWS_ * D3_ * 2);
  __bf16* ctxb   = (__bf16*)alloc((size_t)ROWS_ * D_ * 2);

  int nx = ROWS_ * D_, nw = D3_ * D_, no = D_ * D_;
  f2bf_kernel<<<(nx + 255) / 256, 256, 0, stream>>>(x, xb, nx);
  f2bf_kernel<<<(nw + 255) / 256, 256, 0, stream>>>(in_w, in_wb, nw);
  f2bf_kernel<<<(no + 255) / 256, 256, 0, stream>>>(out_w, out_wb, no);

  state_kernel<<<ROWS_ / 8, 256, 0, stream>>>(x, sp_w, sp_b, st);
  qstats_kernel<<<ROWS_ / 8, 256, 0, stream>>>(st, qmax, qsum);

  qkv_gemm_kernel<<<(ROWS_ / 16) * (D3_ / 64) / 8, 256, 0, stream>>>(xb, in_wb, in_b, qkvb);

  quantum_out_kernel<<<dim3(D_ / 256, S_ / 32, B_), 256, 0, stream>>>(st, qmax, qsum, xb,
                                                                      qw, out);

  attn_kernel<<<dim3(S_ / 64, B_ * H_), 128, 0, stream>>>(qkvb, ctxb);

  outproj_kernel<<<(ROWS_ / 16) * (D_ / 64) / 8, 256, 0, stream>>>(ctxb, out_wb, out_b,
                                                                   qw, out);
}
